// AudioXMMDiTSelfAttention_9345848836634
// MI455X (gfx1250) — matrix operand order repaired and checked by reading, not compiled, not run on hardware
//
#include <hip/hip_runtime.h>
#include <hip/hip_bf16.h>

typedef __attribute__((ext_vector_type(16))) _Float16 v16h;
typedef __attribute__((ext_vector_type(8)))  float    v8f;

#define N_SEQ 2048
#define DIM   1024
#define NHEAD 16
#define HDIM  64
#define NCOL  3072   // 3*DIM

// ---------------------------------------------------------------------------
// CDNA5 async global->LDS copy (CDNA5 ISA, VGLOBAL
// encoding: VDST = LDS byte address VGPR, VADDR = 64-bit global address).
// Tracked by ASYNCcnt; data never touches VGPRs.
// ---------------------------------------------------------------------------
__device__ __forceinline__ void async_load_b128(void* lds_dst,
                                                const void* gsrc) {
  const uint32_t lds_off = (uint32_t)(uintptr_t)lds_dst;
  const uint64_t gaddr   = (uint64_t)(uintptr_t)gsrc;
  asm volatile("global_load_async_to_lds_b128 %0, %1, off"
               :
               : "v"(lds_off), "v"(gaddr)
               : "memory");
}
__device__ __forceinline__ void wait_async0() {
  asm volatile("s_wait_asynccnt 0" ::: "memory");
}

// ---------------------------------------------------------------------------
// Fragment loaders (wave32 WMMA 16x16x32 f16; contiguous k order chosen here, same for A and B)
// A 16x32: lane holds row m=lane&15; lanes 0-15: K=0..15,
//          lanes 16-31: K=16..31 (one contiguous 32B run)
// B 32x16: lane holds col n=lane&15; lanes 0-15: K=0..15, lanes 16-31: K=16..31
//          (LDS laid out so each lane's 16 halfs are one contiguous 32B run)
// C 16x16: lane holds col n=lane&15; vgpr j holds row m = j + 8*(lane>=16)
// ---------------------------------------------------------------------------
__device__ __forceinline__ v16h frag_a_rowmajor(const _Float16* p, int lane) {
  const int kb = (lane & 16) ? 16 : 0;
  v16h a;
#pragma unroll
  for (int e = 0; e < 16; ++e) a[e] = p[kb + e];
  return a;
}
__device__ __forceinline__ v16h frag_k_contig(const _Float16* p) {
  v16h f;
#pragma unroll
  for (int e = 0; e < 16; ++e) f[e] = p[e];
  return f;
}

// ---------------------------------------------------------------------------
// Kernel 0a: x fp32 -> f16 (once; x is re-read 24x by the GEMM)
// ---------------------------------------------------------------------------
__global__ __launch_bounds__(256) void convert_x_kernel(
    const float* __restrict__ x, _Float16* __restrict__ xh) {
  const int i = blockIdx.x * 256 + threadIdx.x;  // float4 units
  float4 v = ((const float4*)x)[i];
  _Float16 h[4] = {(_Float16)v.x, (_Float16)v.y, (_Float16)v.z, (_Float16)v.w};
  *(uint2*)(xh + (size_t)i * 4) = *(const uint2*)h;
}

// ---------------------------------------------------------------------------
// Kernel 0b: W fp32 [k][col] -> Wt f16 [col][k] (LDS-tiled transpose, once)
// ---------------------------------------------------------------------------
__global__ __launch_bounds__(256) void transpose_w_kernel(
    const float* __restrict__ W, _Float16* __restrict__ Wt) {
  __shared__ _Float16 t[64 * 72];
  const int tid = threadIdx.x;
  const int k0  = blockIdx.x * 64;
  const int c0  = blockIdx.y * 64;
#pragma unroll
  for (int i = 0; i < 4; ++i) {
    int idx = tid + i * 256;
    int row = idx >> 4;
    int c4  = idx & 15;
    float4 v = ((const float4*)(W + (size_t)(k0 + row) * NCOL + c0))[c4];
    _Float16* d = &t[(c4 * 4) * 72 + row];
    d[0 * 72] = (_Float16)v.x; d[1 * 72] = (_Float16)v.y;
    d[2 * 72] = (_Float16)v.z; d[3 * 72] = (_Float16)v.w;
  }
  __syncthreads();
#pragma unroll
  for (int i = 0; i < 2; ++i) {
    int idx = tid + i * 256;
    int col = idx >> 3;
    int k8  = (idx & 7) * 8;
    *(uint4*)(Wt + (size_t)(c0 + col) * DIM + k0 + k8) =
        *(const uint4*)(&t[col * 72 + k8]);
  }
}

// ---------------------------------------------------------------------------
// Kernel 1: C[4096,3072] = x @ W + bias (WMMA f16, f32 accum), async
// double-buffered LDS staging. Epilogue de-interleaves col c -> h=c/192,
// d=(c%192)/3, which=c%3: q/k -> f32 [bh][n][d], v -> f16 [bh][d][n].
// ---------------------------------------------------------------------------
#define TSTRIDE1 40  // 32-k tile stride in halfs (80B)
#define T1ELEMS  (128 * TSTRIDE1)

__global__ __launch_bounds__(256) void qkv_gemm_kernel(
    const _Float16* __restrict__ xh, const _Float16* __restrict__ Wt,
    const float* __restrict__ bias,
    float* __restrict__ qf, float* __restrict__ kf, _Float16* __restrict__ vt) {
  __shared__ _Float16 As[2 * T1ELEMS];  // ping-pong [row][k] 128x32
  __shared__ _Float16 Bs[2 * T1ELEMS];  // ping-pong [col][k] 128x32
  __shared__ float    biasS[128];

  const int tid   = threadIdx.x;
  const int lane  = tid & 31;
  const int w     = tid >> 5;
  const int waveM = w & 3;
  const int waveN = w >> 2;
  const int tileM = blockIdx.x * 128;
  const int tileN = blockIdx.y * 128;

  if (tid < 128) biasS[tid] = bias[tileN + tid];

  // issue tile k0 into buffer p (2 x 16B chunks per thread per operand)
  auto issue_tile = [&](int p, int k0) {
#pragma unroll
    for (int i = 0; i < 2; ++i) {
      int idx = tid + i * 256;
      int row = idx >> 2;
      int c8  = (idx & 3) * 8;
      async_load_b128(&As[p * T1ELEMS + row * TSTRIDE1 + c8],
                      xh + (size_t)(tileM + row) * DIM + k0 + c8);
      async_load_b128(&Bs[p * T1ELEMS + row * TSTRIDE1 + c8],
                      Wt + (size_t)(tileN + row) * DIM + k0 + c8);
    }
  };

  v8f acc[2][4] = {};
  issue_tile(0, 0);
  int p = 0;

  for (int k0 = 0; k0 < DIM; k0 += 32) {
    wait_async0();       // this wave's async writes (into buf p) complete
    __syncthreads();     // => all waves' writes into buf p are visible
    if (k0 + 32 < DIM) issue_tile(p ^ 1, k0 + 32);  // prefetch next tile

    const _Float16* A = &As[p * T1ELEMS];
    const _Float16* B = &Bs[p * T1ELEMS];
    const int kb16 = (lane & 16) ? 16 : 0;
    v16h af[2], bf[4];
#pragma unroll
    for (int mi = 0; mi < 2; ++mi)
      af[mi] = frag_a_rowmajor(
          &A[(waveM * 32 + mi * 16 + (lane & 15)) * TSTRIDE1], lane);
#pragma unroll
    for (int ni = 0; ni < 4; ++ni)
      bf[ni] = frag_k_contig(
          &B[(waveN * 64 + ni * 16 + (lane & 15)) * TSTRIDE1 + kb16]);
#pragma unroll
    for (int mi = 0; mi < 2; ++mi)
#pragma unroll
      for (int ni = 0; ni < 4; ++ni)
        acc[mi][ni] = __builtin_amdgcn_wmma_f32_16x16x32_f16(
            false, af[mi], false, bf[ni], (short)0, acc[mi][ni], false, false);
    p ^= 1;
  }

  // Epilogue: bias + de-interleave scatter
#pragma unroll
  for (int mi = 0; mi < 2; ++mi) {
#pragma unroll
    for (int ni = 0; ni < 4; ++ni) {
      const int colL  = waveN * 64 + ni * 16 + (lane & 15);
      const int col   = tileN + colL;
      const int h     = col / 192;
      const int rrem  = col - h * 192;
      const int d     = rrem / 3;
      const int which = rrem - d * 3;
      const int rowB  = tileM + waveM * 32 + mi * 16 + ((lane & 16) ? 8 : 0);
      const float bv  = biasS[colL];
#pragma unroll
      for (int j = 0; j < 8; ++j) {
        const int row = rowB + j;  // global row = b*N + n
        const int b   = row >> 11;
        const int n   = row & (N_SEQ - 1);
        const int bh  = b * NHEAD + h;
        const float val = acc[mi][ni][j] + bv;
        if (which == 0)
          qf[((size_t)bh * N_SEQ + n) * HDIM + d] = val;
        else if (which == 1)
          kf[((size_t)bh * N_SEQ + n) * HDIM + d] = val;
        else
          vt[((size_t)bh * HDIM + d) * N_SEQ + n] = (_Float16)val;
      }
    }
  }
}

// ---------------------------------------------------------------------------
// Kernel 2: RMSNorm (eps=1e-6, no weight) + RoPE, fp32 math -> f16 output.
// One wave per (b,h,n) row of 64; lane L owns pair (2L, 2L+1).
// ---------------------------------------------------------------------------
__global__ __launch_bounds__(256) void norm_rope_kernel(
    const float* __restrict__ qf, const float* __restrict__ kf,
    const float* __restrict__ rot,
    _Float16* __restrict__ qh, _Float16* __restrict__ kh) {
  const int lane = threadIdx.x & 31;
  const int w    = threadIdx.x >> 5;
  int r = blockIdx.x * 8 + w;

  const float* src;
  _Float16*    dst;
  if (r < 65536) { src = qf; dst = qh; }
  else           { src = kf; dst = kh; r -= 65536; }

  const int n = r & (N_SEQ - 1);
  const float* p = src + (size_t)r * HDIM + lane * 2;
  float t0 = p[0], t1 = p[1];

  float ss = t0 * t0 + t1 * t1;
#pragma unroll
  for (int m = 16; m >= 1; m >>= 1) ss += __shfl_xor(ss, m, 32);
  const float s = rsqrtf(ss * (1.0f / 64.0f) + 1e-6f);
  t0 *= s; t1 *= s;

  const float4 rv = *(const float4*)(rot + (size_t)n * 128 + lane * 4);
  const float o0 = rv.x * t0 + rv.y * t1;  // cos*t0 - sin*t1
  const float o1 = rv.z * t0 + rv.w * t1;  // sin*t0 + cos*t1

  _Float16* q = dst + (size_t)r * HDIM + lane * 2;
  q[0] = (_Float16)o0;
  q[1] = (_Float16)o1;
}

// ---------------------------------------------------------------------------
// Kernel 3: flash attention. Block = (b,h, 128-query tile), 8 waves x 16 rows.
// K/V tiles async-prefetched into ping-pong LDS; Q/K LDS as [row][d],
// V as [d][key] (pre-transposed vt) so all WMMA operands are contiguous runs.
// ---------------------------------------------------------------------------
#define TS 80                 // padded LDS row stride (halfs): 160B
#define KVELEMS (64 * TS)     // one 64-row tile

__global__ __launch_bounds__(256) void attn_kernel(
    const _Float16* __restrict__ qh, const _Float16* __restrict__ kh,
    const _Float16* __restrict__ vt, float* __restrict__ out) {
  __shared__ _Float16 Ks[2 * KVELEMS];   // ping-pong [key][d]
  __shared__ _Float16 Vs[2 * KVELEMS];   // ping-pong [d][key]
  __shared__ _Float16 Ps[8][16 * TS];    // per-wave [qrow][key]

  const int tid   = threadIdx.x;
  const int lane  = tid & 31;
  const int w     = tid >> 5;
  const int qTile = blockIdx.x * 128;
  const int bh    = blockIdx.y;  // b*16 + h
  const int b     = bh >> 4;
  const int h     = bh & 15;

  const _Float16* qb = qh + (size_t)bh * N_SEQ * HDIM;
  const _Float16* kb = kh + (size_t)bh * N_SEQ * HDIM;
  const _Float16* vb = vt + (size_t)bh * HDIM * N_SEQ;

  // Stage Q tile (128x64) synchronously into the K ping-pong area, grab
  // per-wave A fragments, then release the buffer to the pipeline.
#pragma unroll
  for (int i = 0; i < 4; ++i) {
    int idx = tid + i * 256;  // 1024 x 8-half chunks
    int row = idx >> 3;
    int c8  = (idx & 7) * 8;
    *(uint4*)(&Ks[row * TS + c8]) =
        *(const uint4*)(&qb[(size_t)(qTile + row) * HDIM + c8]);
  }
  __syncthreads();
  v16h aq[2];
#pragma unroll
  for (int kc = 0; kc < 2; ++kc)
    aq[kc] = frag_a_rowmajor(&Ks[(w * 16 + (lane & 15)) * TS + kc * 32], lane);
  __syncthreads();

  // async prefetch of one 64-key K or V tile: 2 x 16B chunks per thread
  auto issue_kv = [&](int p, int kt) {
#pragma unroll
    for (int i = 0; i < 2; ++i) {
      int idx = tid + i * 256;  // 512 chunks
      int row = idx >> 3;
      int c8  = (idx & 7) * 8;
      async_load_b128(&Ks[p * KVELEMS + row * TS + c8],
                      &kb[(size_t)(kt + row) * HDIM + c8]);
      async_load_b128(&Vs[p * KVELEMS + row * TS + c8],
                      &vb[(size_t)row * N_SEQ + kt + c8]);
    }
  };

  float mrow[8], lrow[8];
  v8f accO[4] = {};
#pragma unroll
  for (int j = 0; j < 8; ++j) { mrow[j] = -3.0e38f; lrow[j] = 0.0f; }

  const int kb16 = (lane & 16) ? 16 : 0;
  issue_kv(0, 0);
  int p = 0;

  for (int kt = 0; kt < N_SEQ; kt += 64) {
    wait_async0();
    __syncthreads();
    if (kt + 64 < N_SEQ) issue_kv(p ^ 1, kt + 64);  // prefetch next tile

    const _Float16* K = &Ks[p * KVELEMS];
    const _Float16* V = &Vs[p * KVELEMS];

    // S = Q K^T * scale
    float sblk[4][8];
#pragma unroll
    for (int nb = 0; nb < 4; ++nb) {
      v8f s = {};
#pragma unroll
      for (int kc = 0; kc < 2; ++kc) {
        v16h bt = frag_k_contig(
            &K[(nb * 16 + (lane & 15)) * TS + kc * 32 + kb16]);
        s = __builtin_amdgcn_wmma_f32_16x16x32_f16(false, aq[kc], false, bt,
                                                   (short)0, s, false, false);
      }
#pragma unroll
      for (int j = 0; j < 8; ++j) sblk[nb][j] = s[j] * 0.125f;  // 1/sqrt(64)
    }

    // Online softmax (row reductions across the 16-lane column groups)
    float alpha[8];
#pragma unroll
    for (int j = 0; j < 8; ++j) {
      float mloc = fmaxf(fmaxf(sblk[0][j], sblk[1][j]),
                         fmaxf(sblk[2][j], sblk[3][j]));
#pragma unroll
      for (int m = 8; m >= 1; m >>= 1)
        mloc = fmaxf(mloc, __shfl_xor(mloc, m, 16));
      const float mnew = fmaxf(mrow[j], mloc);
      alpha[j] = __expf(mrow[j] - mnew);
      float rs = 0.0f;
#pragma unroll
      for (int nb = 0; nb < 4; ++nb) {
        const float pv = __expf(sblk[nb][j] - mnew);
        sblk[nb][j] = pv;
        rs += pv;
      }
#pragma unroll
      for (int m = 8; m >= 1; m >>= 1) rs += __shfl_xor(rs, m, 16);
      lrow[j] = lrow[j] * alpha[j] + rs;
      mrow[j] = mnew;
    }
#pragma unroll
    for (int db = 0; db < 4; ++db)
#pragma unroll
      for (int j = 0; j < 8; ++j) accO[db][j] *= alpha[j];

    // C-layout P -> per-wave LDS -> A-layout fragments
    {
      const int col   = lane & 15;
      const int rbase = (lane & 16) ? 8 : 0;
#pragma unroll
      for (int nb = 0; nb < 4; ++nb)
#pragma unroll
        for (int j = 0; j < 8; ++j)
          Ps[w][(rbase + j) * TS + nb * 16 + col] = (_Float16)sblk[nb][j];
    }
    asm volatile("s_wait_dscnt 0" ::: "memory");  // wave-local DS RAW fence
    v16h ap[2];
#pragma unroll
    for (int kc = 0; kc < 2; ++kc)
      ap[kc] = frag_a_rowmajor(&Ps[w][(lane & 15) * TS + kc * 32], lane);

    // O += P V
#pragma unroll
    for (int db = 0; db < 4; ++db) {
#pragma unroll
      for (int kc = 0; kc < 2; ++kc) {
        v16h bv = frag_k_contig(
            &V[(db * 16 + (lane & 15)) * TS + kc * 32 + kb16]);
        accO[db] = __builtin_amdgcn_wmma_f32_16x16x32_f16(
            false, ap[kc], false, bv, (short)0, accO[db], false, false);
      }
    }
    __syncthreads();  // all waves done reading buf p before it is re-filled
    p ^= 1;
  }

  // Normalize and write out[b, n, h*64 + d] (fp32)
  const int col   = lane & 15;
  const int rbase = (lane & 16) ? 8 : 0;
#pragma unroll
  for (int db = 0; db < 4; ++db) {
#pragma unroll
    for (int j = 0; j < 8; ++j) {
      const int n = qTile + w * 16 + rbase + j;
      const int d = db * 16 + col;
      out[((size_t)b * N_SEQ + n) * DIM + h * HDIM + d] =
          accO[db][j] / lrow[j];
    }
  }
}

// ---------------------------------------------------------------------------
// Launch
// ---------------------------------------------------------------------------
extern "C" void kernel_launch(void* const* d_in, const int* in_sizes, int n_in,
                              void* d_out, int out_size, void* d_ws,
                              size_t ws_size, hipStream_t stream) {
  const float* x    = (const float*)d_in[0];
  const float* W    = (const float*)d_in[1];
  const float* bias = (const float*)d_in[2];
  const float* rot  = (const float*)d_in[3];
  float*       out  = (float*)d_out;

  // Workspace layout (70MB total):
  char* ws = (char*)d_ws;
  float*    qf = (float*)(ws);                     // 16MB f32 [bh][n][d]
  float*    kf = (float*)(ws + (16ull << 20));     // 16MB f32 [bh][n][d]
  _Float16* qh = (_Float16*)(ws + (32ull << 20));  //  8MB f16 [bh][n][d]
  _Float16* kh = (_Float16*)(ws + (40ull << 20));  //  8MB f16 [bh][n][d]
  _Float16* vt = (_Float16*)(ws + (48ull << 20));  //  8MB f16 [bh][d][n]
  _Float16* Wt = (_Float16*)(ws + (56ull << 20));  //  6MB f16 [col][k]
  _Float16* xh = (_Float16*)(ws + (62ull << 20));  //  8MB f16 [row][k]

  convert_x_kernel<<<8192, 256, 0, stream>>>(x, xh);
  transpose_w_kernel<<<dim3(16, 48), 256, 0, stream>>>(W, Wt);
  qkv_gemm_kernel<<<dim3(32, 24), 256, 0, stream>>>(xh, Wt, bias, qf, kf, vt);
  norm_rope_kernel<<<16384, 256, 0, stream>>>(qf, kf, rot, qh, kh);
  attn_kernel<<<dim3(16, 32), 256, 0, stream>>>(qh, kh, vt, out);
}
